// SAGE_11364483465234
// MI455X (gfx1250) — matrix operand order repaired and checked by reading, not compiled, not run on hardware
//
#include <hip/hip_runtime.h>
#include <cstdint>
#include <cstddef>

typedef __attribute__((ext_vector_type(16))) _Float16 v16h;
typedef __attribute__((ext_vector_type(8)))  _Float16 v8h;
typedef __attribute__((ext_vector_type(8)))  float    v8f;

#define TPB 256
#define MTILES 4   // M-tiles per wave sharing one B-operand

static __host__ int nblk(long n) { return (int)((n + TPB - 1) / TPB); }

// ---------------- utility kernels ----------------

__global__ void k_zero_f32(float* __restrict__ p, long n) {
  long i = (long)blockIdx.x * blockDim.x + threadIdx.x;
  if (i < n) p[i] = 0.0f;
}

__global__ void k_deg(const int* __restrict__ erow, long E, float* __restrict__ deg) {
  long i = (long)blockIdx.x * blockDim.x + threadIdx.x;
  if (i < E) unsafeAtomicAdd(&deg[erow[i]], 1.0f);
}

__global__ void k_invdeg(const float* __restrict__ deg, float* __restrict__ inv, long n) {
  long i = (long)blockIdx.x * blockDim.x + threadIdx.x;
  if (i < n) { float d = deg[i]; inv[i] = d > 0.0f ? 1.0f / d : 0.0f; }
}

// ---------------- sparse mean-aggregation: one wave per edge ----------------
// agg[row][:] += act[col][:]; each of 32 lanes moves a float2 (256B/edge).
// act (25.6MB) is L2-resident on a 192MB-L2 part, so the random gather hits L2.

__global__ void k_scatter(const float* __restrict__ act,
                          const int* __restrict__ erow, const int* __restrict__ ecol,
                          float* __restrict__ agg, long E) {
  long gid  = (long)blockIdx.x * blockDim.x + threadIdx.x;
  long e    = gid >> 5;
  if (e >= E) return;
  int  lane = (int)(gid & 31);
  if (e + 2048 < E) __builtin_prefetch(ecol + e + 2048, 0, 0);   // global_prefetch_b8
  int r = erow[e];
  int c = ecol[e];
  const float2 v = *(const float2*)(act + (size_t)c * 64 + lane * 2);
  float* dst = agg + (size_t)r * 64 + lane * 2;
  unsafeAtomicAdd(dst,     v.x);   // global_atomic_add_f32
  unsafeAtomicAdd(dst + 1, v.y);
}

// ------- scale agg by inv_deg + convert to f16; zero-fill row padding -------

__global__ void k_convert(const float* __restrict__ agg, const float* __restrict__ act,
                          const float* __restrict__ inv,
                          _Float16* __restrict__ aggH, _Float16* __restrict__ actH,
                          long n64, long nPad64) {
  long i = (long)blockIdx.x * blockDim.x + threadIdx.x;
  if (i >= nPad64) return;
  if (i < n64) {
    aggH[i] = (_Float16)(agg[i] * inv[i >> 6]);
    actH[i] = (_Float16)act[i];
  } else {                       // pad rows up to rowTiles*16: defined zeros
    aggH[i] = (_Float16)0.0f;
    actH[i] = (_Float16)0.0f;
  }
}

// ---------------- pack [Wl;Wr] (K=128 x Fo) into the WMMA B-lane layout -----
// B 32x16 chunk: lanes 0-15 hold K=kc+0..7,16..23, lanes 16-31 hold K=kc+8..15,24..31,
// lane's 16 halfs contiguous -> GEMM does one 32B load per chunk.
// Columns >= Fo are zero-padded so the GEMM needs no column guard.

__global__ void k_packW(const float* __restrict__ Wl, const float* __restrict__ Wr,
                        int Fo, int nTiles, _Float16* __restrict__ packed) {
  int total = 4 * nTiles * 32 * 16;
  int idx = blockIdx.x * blockDim.x + threadIdx.x;
  if (idx >= total) return;
  int h = idx & 15;
  int L = (idx >> 4) & 31;
  int t = (idx >> 9) % nTiles;
  int c = (idx >> 9) / nTiles;          // k-chunk 0..3 (0,1 -> Wl; 2,3 -> Wr)
  int kb2 = (L < 16) ? 0 : 8;
  int kk  = (c & 1) * 32 + ((h < 8) ? kb2 + h : 16 + kb2 + (h - 8));     // 0..63 within Wl or Wr
  int n   = t * 16 + (L & 15);
  const float* W = (c < 2) ? Wl : Wr;
  float v = (n < Fo) ? W[(size_t)kk * Fo + n] : 0.0f;
  packed[idx] = (_Float16)v;
}

__global__ void k_bias(const float* __restrict__ bl, const float* __restrict__ br,
                       int Fo, float* __restrict__ biasC) {
  int i = threadIdx.x;
  if (i < 64) biasC[i] = (i < Fo) ? (bl[i] + br[i]) : 0.0f;
}

// ---------------- fused dual-GEMM: out = [aggH|actH] @ packedB + bias -------
// One wave computes MTILES consecutive 16x16 tiles in the same column strip,
// reusing the 4 hoisted B chunks. K = 128 = 4 chunks of 32, f16->f32 WMMA.
// All row buffers are padded to rowTiles*16 -> fully unconditional stores with
// compile-time stride LD (immediate-offset global_store_b32).

template <int LD, bool RELU>
__global__ void k_gemm(const _Float16* __restrict__ aggH, const _Float16* __restrict__ actH,
                       const _Float16* __restrict__ pB, const float* __restrict__ biasC,
                       float* __restrict__ out, int nTiles, int rowTiles) {
  int mtGroups = (rowTiles + MTILES - 1) / MTILES;
  int w = blockIdx.x * (blockDim.x >> 5) + (threadIdx.x >> 5);
  if (w >= mtGroups * nTiles) return;           // wave-uniform: EXEC stays all-ones
  int g    = w / nTiles;
  int nt   = w % nTiles;
  int lane = threadIdx.x & 31;
  int mrow = lane & 15;
  int kb   = (lane < 16) ? 0 : 8;               // ISA 16-bit A-layout lane split

  // hoist B: 4 chunks x v16h, reused across MTILES row tiles
  v16h b[4];
#pragma unroll
  for (int c = 0; c < 4; ++c)
    b[c] = *(const v16h*)(pB + (((long)c * nTiles + nt) * 32 + lane) * 16);

  int   ncol = nt * 16 + (lane & 15);
  float bias = biasC[ncol & 63];

  for (int mi = 0; mi < MTILES; ++mi) {
    int mt = g * MTILES + mi;
    if (mt >= rowTiles) break;                  // wave-uniform
    long r = (long)mt * 16 + mrow;

    v8f acc = {};
#pragma unroll
    for (int c = 0; c < 4; ++c) {
      const _Float16* src = (c < 2) ? aggH : actH;
      int kc = (c & 1) * 32;
      const _Float16* ap = src + r * 64 + kc + kb;
      v8h lo = *(const v8h*)(ap);               // K = kc+kb .. +7
      v8h hi = *(const v8h*)(ap + 16);          // K = kc+kb+16 .. +23
      v16h a;
#pragma unroll
      for (int i = 0; i < 8; ++i) { a[i] = lo[i]; a[i + 8] = hi[i]; }
      acc = __builtin_amdgcn_wmma_f32_16x16x32_f16(false, a, false, b[c],
                                                   (short)0, acc, false, false);
    }

    long   rbase = (long)mt * 16 + ((lane < 16) ? 0 : 8);
    float* dst   = out + rbase * LD + ncol;     // one base; 8 imm-offset stores
#pragma unroll
    for (int i = 0; i < 8; ++i) {
      float v = acc[i] + bias;
      if (RELU) v = v > 0.0f ? v : 0.0f;
      dst[i * LD] = v;
    }
  }
}

// ---------------- log-softmax over Fo cols (stride ld): one wave per row ----

__global__ void k_logsoftmax(const float* __restrict__ in, float* __restrict__ out,
                             long n, int ld, int Fo) {
  long row = (long)blockIdx.x * (blockDim.x >> 5) + (threadIdx.x >> 5);
  if (row >= n) return;
  int lane = threadIdx.x & 31;
  const float NEG = -3.4e38f;
  float v0 = (lane      < Fo) ? in[row * ld + lane]      : NEG;
  float v1 = (lane + 32 < Fo) ? in[row * ld + lane + 32] : NEG;
  float m = fmaxf(v0, v1);
  for (int o = 16; o > 0; o >>= 1) m = fmaxf(m, __shfl_xor(m, o, 32));
  float s = ((lane      < Fo) ? expf(v0 - m) : 0.0f) +
            ((lane + 32 < Fo) ? expf(v1 - m) : 0.0f);
  for (int o = 16; o > 0; o >>= 1) s += __shfl_xor(s, o, 32);
  float l = logf(s);
  if (lane      < Fo) out[row * Fo + lane]      = v0 - m - l;
  if (lane + 32 < Fo) out[row * Fo + lane + 32] = v1 - m - l;
}

// ---------------- launch ----------------

extern "C" void kernel_launch(void* const* d_in, const int* in_sizes, int n_in,
                              void* d_out, int out_size, void* d_ws, size_t ws_size,
                              hipStream_t stream) {
  const float* x    = (const float*)d_in[0];
  const int*   erow = (const int*)d_in[1];
  const int*   ecol = (const int*)d_in[2];
  const float* Wl[3] = { (const float*)d_in[5],  (const float*)d_in[9],  (const float*)d_in[13] };
  const float* bl[3] = { (const float*)d_in[6],  (const float*)d_in[10], (const float*)d_in[14] };
  const float* Wr[3] = { (const float*)d_in[7],  (const float*)d_in[11], (const float*)d_in[15] };
  const float* br[3] = { (const float*)d_in[8],  (const float*)d_in[12], (const float*)d_in[16] };

  const long n        = in_sizes[0] / 64;        // N nodes, F_IN = 64
  const long E        = in_sizes[1];
  const int  rowTiles = (int)((n + 15) / 16);
  const long nPad     = (long)rowTiles * 16;     // padded node count
  const long n64      = n * 64;
  const long nPad64   = nPad * 64;

  // workspace carve-out (256B aligned slices), all node buffers row-padded
  char* ws = (char*)d_ws;
  size_t off = 0;
  auto take = [&](size_t bytes) -> char* {
    off = (off + 255) & ~(size_t)255;
    char* p = ws + off;
    off += bytes;
    return p;
  };
  float*    bufA  = (float*)take(nPad64 * sizeof(float));      // ~25.6 MB
  float*    bufB  = (float*)take(nPad64 * sizeof(float));      // ~25.6 MB
  _Float16* aggH  = (_Float16*)take(nPad64 * sizeof(_Float16));
  _Float16* actH  = (_Float16*)take(nPad64 * sizeof(_Float16));
  float*    deg   = (float*)take(n * sizeof(float));
  float*    invd  = (float*)take(n * sizeof(float));
  _Float16* pW[3];
  float*    pb[3];
  for (int l = 0; l < 3; ++l) {
    pW[l] = (_Float16*)take(4 * 4 * 32 * 16 * sizeof(_Float16));  // worst case nTiles=4
    pb[l] = (float*)take(64 * sizeof(float));
  }
  (void)ws_size;

  const int nTilesL[3] = {4, 4, 3};
  const int FoL[3]     = {64, 64, 40};

  // in-degree -> inverse
  k_zero_f32<<<nblk(n), TPB, 0, stream>>>(deg, n);
  k_deg<<<nblk(E), TPB, 0, stream>>>(erow, E, deg);
  k_invdeg<<<nblk(n), TPB, 0, stream>>>(deg, invd, n);

  // pre-pack weights + biases into WMMA-B layout (tiny, once per launch)
  for (int l = 0; l < 3; ++l) {
    int total = 4 * nTilesL[l] * 32 * 16;
    k_packW<<<nblk(total), TPB, 0, stream>>>(Wl[l], Wr[l], FoL[l], nTilesL[l], pW[l]);
    k_bias<<<1, 64, 0, stream>>>(bl[l], br[l], FoL[l], pb[l]);
  }

  // layer schedule: L0 act=x, agg=bufB, out=bufA (ld 64, relu)
  //                 L1 act=bufA, agg=bufB, out=bufB (ld 64, relu)
  //                 L2 act=bufB, agg=bufA, out=bufA (ld 48, linear)
  const float* actf[3] = { x, bufA, bufB };
  float*       aggf[3] = { bufB, bufB, bufA };
  float*       outf[3] = { bufA, bufB, bufA };

  const long mtGroups = (rowTiles + MTILES - 1) / MTILES;

  for (int l = 0; l < 3; ++l) {
    k_zero_f32<<<nblk(n64), TPB, 0, stream>>>(aggf[l], n64);
    k_scatter<<<nblk(E * 32), TPB, 0, stream>>>(actf[l], erow, ecol, aggf[l], E);
    k_convert<<<nblk(nPad64), TPB, 0, stream>>>(aggf[l], actf[l], invd, aggH, actH,
                                                n64, nPad64);
    long waves = mtGroups * nTilesL[l];
    if (l != 2)
      k_gemm<64, true><<<nblk(waves * 32), TPB, 0, stream>>>(aggH, actH, pW[l], pb[l],
                                                             outf[l], nTilesL[l], rowTiles);
    else
      k_gemm<48, false><<<nblk(waves * 32), TPB, 0, stream>>>(aggH, actH, pW[l], pb[l],
                                                              outf[l], nTilesL[l], rowTiles);
  }

  // final log-softmax (8 waves per block, one row per wave), padded-stride input
  k_logsoftmax<<<nblk(n * 32), TPB, 0, stream>>>(outf[2], (float*)d_out, n, 48, 40);
}
